// GCN_6966436954284
// MI455X (gfx1250) — matrix operand order repaired and checked by reading, not compiled, not run on hardware
//
#include <hip/hip_runtime.h>
#include <hip/hip_bf16.h>

// ---------------------------------------------------------------------------
// GCN forward for MI455X (gfx1250, wave32, WMMA).
// Dense layers: v_wmma_f32_16x16x32_bf16, B-operands staged in LDS in
// fragment-native order so each fragment is one 32B ds_load_b128 pair.
// Aggregation: L2-resident scatter-add with global_atomic_add_f32.
// ---------------------------------------------------------------------------

typedef __attribute__((ext_vector_type(16))) __bf16 v16bf;
typedef __attribute__((ext_vector_type(8)))  float  v8f;

#define NNODES 100000
#define NEDGES 1600000
#define EPSBN  1e-5f

// ---------------- graph normalization kernels ----------------

__global__ void k_deg_init(float* __restrict__ deg, int n) {
  int i = blockIdx.x * blockDim.x + threadIdx.x;
  if (i < n) deg[i] = 1.0f;                       // self-loop weight
}

__global__ void k_deg_accum(const long long* __restrict__ dst,
                            const float* __restrict__ w,
                            float* __restrict__ deg, int e) {
  int i = blockIdx.x * blockDim.x + threadIdx.x;
  if (i < e) atomicAdd(&deg[dst[i]], w[i]);
}

__global__ void k_dinv(const float* __restrict__ deg, float* __restrict__ dinv, int n) {
  int i = blockIdx.x * blockDim.x + threadIdx.x;
  if (i < n) dinv[i] = rsqrtf(deg[i]);
}

__global__ void k_norm(const long long* __restrict__ src,
                       const long long* __restrict__ dst,
                       const float* __restrict__ w,
                       const float* __restrict__ dinv,
                       float* __restrict__ norm, int e) {
  int i = blockIdx.x * blockDim.x + threadIdx.x;
  if (i < e) norm[i] = dinv[src[i]] * w[i] * dinv[dst[i]];
}

// ---------------- elementwise kernels ----------------

__global__ void k_f32_to_bf16(const float* __restrict__ in, __bf16* __restrict__ out,
                              int n) {
  int i = blockIdx.x * blockDim.x + threadIdx.x;
  if (i < n) out[i] = (__bf16)in[i];
}

// one wave per edge: agg[dst,f] += t[src,f] * norm[e]
template <int F>
__global__ void k_scatter(const float* __restrict__ t,
                          const long long* __restrict__ src,
                          const long long* __restrict__ dst,
                          const float* __restrict__ norm,
                          float* __restrict__ agg, int e) {
  int gw   = (blockIdx.x * blockDim.x + threadIdx.x) >> 5;
  int lane = threadIdx.x & 31;
  if (gw >= e) return;
  long long s = src[gw], d = dst[gw];
  float nv = norm[gw];
  const float* tp = t + (size_t)s * F;
  float* ap = agg + (size_t)d * F;
#pragma unroll
  for (int f = lane; f < F; f += 32) atomicAdd(&ap[f], tp[f] * nv);
}

template <int F>
__global__ void k_bn_relu_bf16(const float* __restrict__ agg,
                               const float* __restrict__ g,
                               const float* __restrict__ b,
                               const float* __restrict__ m,
                               const float* __restrict__ v,
                               __bf16* __restrict__ out, int n) {
  int i = blockIdx.x * blockDim.x + threadIdx.x;
  if (i >= n * F) return;
  int f = i % F;
  float x = (agg[i] - m[f]) * rsqrtf(v[f] + EPSBN) * g[f] + b[f];
  out[i] = (__bf16)fmaxf(x, 0.0f);
}

// ---------------- WMMA bf16 GEMM with fused epilogues ----------------
// T[n,FOUT] = A[n,KDIM] @ W[KDIM,FOUT]
// Block = 256 threads = 8 waves; each wave owns one 16-row M-tile and loops
// over FOUT/16 N-tiles. W (f32) is converted to bf16 and permuted into
// B-fragment-native order in LDS: Wswz[frag][lane] = 16 contiguous bf16
// elements -> each fragment load is one aligned 32B LDS read.
//
// Fragment layouts per CDNA5 ISA 7.12.2 (wave32, 16-bit):
//   A (16x32): lane group g=lane>>4, row M=lane&15,
//              element e holds K = e + 8*g + (e>=8 ? 8 : 0)   (two 16B runs)
//   B (32x16): col N=lane&15, element e holds K = e + 8*g + (e>=8 ? 8 : 0), same as A
//   C (16x16 f32, v8f): element rr holds M = rr + 8*g, N = lane&15
//
// MODE 0: store f32 T
// MODE 1: store f32 T and agg = T * dinv[row]^2 + bias[col]   (conv init)
// MODE 2: store bf16 relu(T + bias[col])                       (lin_in)

template <int KDIM, int FOUT, int MODE>
__global__ __launch_bounds__(256)
void k_gemm_bf16(const __bf16* __restrict__ A, const float* __restrict__ W,
                 float* __restrict__ T, __bf16* __restrict__ OB,
                 const float* __restrict__ dinv, const float* __restrict__ bias,
                 float* __restrict__ agg, int nrows) {
  constexpr int NTN = FOUT / 16;
  constexpr int NKC = KDIM / 32;
  constexpr int NFRAG = NKC * NTN;
  __shared__ v16bf Wswz[NFRAG * 32];

  // swizzle-fill: W[k, col] (f32) -> bf16 fragment order (frag, lane, elem).
  // Each i packs one element pair (e=2d, 2d+1) into one LDS dword.
  {
    unsigned* Wd = (unsigned*)Wswz;
    for (int i = threadIdx.x; i < NFRAG * 32 * 8; i += 256) {
      int fidx = i >> 8;                  // / (32 lanes * 8 dwords)
      int rem  = i & 255;
      int l = rem >> 3, d = rem & 7;
      int kc = fidx / NTN, nt = fidx - kc * NTN;
      int k   = kc * 32 + ((l >> 4) << 3) + 2 * d + ((d >> 2) << 3);   // K = e + 8*group + (e>=8 ? 8 : 0)
      int col = nt * 16 + (l & 15);
      unsigned short u0 = __builtin_bit_cast(unsigned short, (__bf16)W[(size_t)k * FOUT + col]);
      unsigned short u1 = __builtin_bit_cast(unsigned short, (__bf16)W[(size_t)(k + 1) * FOUT + col]);
      Wd[i] = (unsigned)u0 | ((unsigned)u1 << 16);
    }
  }
  __syncthreads();

  const int lane  = threadIdx.x & 31;
  const int wave  = threadIdx.x >> 5;
  const int mtile = blockIdx.x * 8 + wave;
  if (mtile * 16 >= nrows) return;        // wave-uniform exit, EXEC stays all-1
  const int row0 = mtile * 16;
  const int g    = lane >> 4;
  const int r    = lane & 15;

  // Preload all A fragments for this wave's 16 rows: 2x 16B vector loads each.
  v16bf afrag[NKC];
  const __bf16* arow = A + (size_t)(row0 + r) * KDIM;
#pragma unroll
  for (int kc = 0; kc < NKC; ++kc) {
    uint4 lo = *(const uint4*)(arow + kc * 32 + 8 * g);        // K = 8g .. 8g+7
    uint4 hi = *(const uint4*)(arow + kc * 32 + 16 + 8 * g);   // K = 16+8g .. 23+8g
    uint4* ap = (uint4*)&afrag[kc];
    ap[0] = lo;
    ap[1] = hi;
  }

  float sn[8];
  if constexpr (MODE == 1) {
#pragma unroll
    for (int rr = 0; rr < 8; ++rr) {
      float dd = dinv[row0 + rr + 8 * g];
      sn[rr] = dd * dd;
    }
  }

#pragma unroll
  for (int nt = 0; nt < NTN; ++nt) {
    v8f acc = {};
#pragma unroll
    for (int kc = 0; kc < NKC; ++kc) {
      v16bf bfrag = Wswz[(kc * NTN + nt) * 32 + lane];   // one 32B LDS read
      acc = __builtin_amdgcn_wmma_f32_16x16x32_bf16(
          /*neg_a=*/false, afrag[kc], /*neg_b=*/false, bfrag,
          /*c_mod=*/(short)0, acc, /*reuse_a=*/false, /*reuse_b=*/false);
    }
    const int col = nt * 16 + r;
    if constexpr (MODE == 0) {
      float* tp = T + (size_t)row0 * FOUT + col;
#pragma unroll
      for (int rr = 0; rr < 8; ++rr) tp[(size_t)(rr + 8 * g) * FOUT] = acc[rr];
    } else if constexpr (MODE == 1) {
      float bv = bias[col];
      float* tp = T + (size_t)row0 * FOUT + col;
      float* gp = agg + (size_t)row0 * FOUT + col;
#pragma unroll
      for (int rr = 0; rr < 8; ++rr) {
        float v = acc[rr];
        tp[(size_t)(rr + 8 * g) * FOUT] = v;
        gp[(size_t)(rr + 8 * g) * FOUT] = v * sn[rr] + bv;
      }
    } else {
      float bv = bias[col];
      __bf16* op = OB + (size_t)row0 * FOUT + col;
#pragma unroll
      for (int rr = 0; rr < 8; ++rr)
        op[(size_t)(rr + 8 * g) * FOUT] = (__bf16)fmaxf(acc[rr] + bv, 0.0f);
    }
  }
}

// ---------------- final log-softmax over 16 logits ----------------

__global__ void k_logsoftmax16(const float* __restrict__ t,
                               const float* __restrict__ bo,
                               float* __restrict__ out, int n) {
  int i = blockIdx.x * blockDim.x + threadIdx.x;
  if (i >= n) return;
  float v[16];
  float mx = -3.4e38f;
#pragma unroll
  for (int c = 0; c < 16; ++c) {
    v[c] = t[(size_t)i * 16 + c] + bo[c];
    mx = fmaxf(mx, v[c]);
  }
  float s = 0.0f;
#pragma unroll
  for (int c = 0; c < 16; ++c) s += __expf(v[c] - mx);
  float lse = mx + __logf(s);
#pragma unroll
  for (int c = 0; c < 16; ++c) out[(size_t)i * 16 + c] = v[c] - lse;
}

// ---------------------------------------------------------------------------

static inline int cdiv(long long a, long long b) { return (int)((a + b - 1) / b); }

extern "C" void kernel_launch(void* const* d_in, const int* in_sizes, int n_in,
                              void* d_out, int out_size, void* d_ws, size_t ws_size,
                              hipStream_t stream) {
  const float*     x        = (const float*)d_in[0];
  const long long* ei       = (const long long*)d_in[1];   // int64 [2, E]
  const float*     ew       = (const float*)d_in[2];
  const float*     lin_in_w = (const float*)d_in[3];
  const float*     lin_in_b = (const float*)d_in[4];
  const float*     conv1_w  = (const float*)d_in[5];
  const float*     conv1_b  = (const float*)d_in[6];
  const float*     bn1_g    = (const float*)d_in[7];
  const float*     bn1_b    = (const float*)d_in[8];
  const float*     bn1_m    = (const float*)d_in[9];
  const float*     bn1_v    = (const float*)d_in[10];
  const float*     conv2_w  = (const float*)d_in[11];
  const float*     conv2_b  = (const float*)d_in[12];
  const float*     bn2_g    = (const float*)d_in[13];
  const float*     bn2_b    = (const float*)d_in[14];
  const float*     bn2_m    = (const float*)d_in[15];
  const float*     bn2_v    = (const float*)d_in[16];
  const float*     conv3_w  = (const float*)d_in[17];
  const float*     conv3_b  = (const float*)d_in[18];
  const float*     bn3_g    = (const float*)d_in[19];
  const float*     bn3_b    = (const float*)d_in[20];
  const float*     bn3_m    = (const float*)d_in[21];
  const float*     bn3_v    = (const float*)d_in[22];
  const float*     lout_w   = (const float*)d_in[23];
  const float*     lout_b   = (const float*)d_in[24];
  float* out = (float*)d_out;

  const long long* src = ei;
  const long long* dst = ei + NEDGES;

  // ---- workspace carve-up (everything written before read) ----
  char* ws = (char*)d_ws;
  size_t off = 0;
  auto carve = [&](size_t bytes) -> char* {
    char* p = ws + off;
    off = (off + bytes + 255) & ~(size_t)255;
    return p;
  };
  float*  deg  = (float*)carve((size_t)NNODES * 4);
  float*  dinv = (float*)carve((size_t)NNODES * 4);
  float*  norm = (float*)carve((size_t)NEDGES * 4);
  __bf16* hbA  = (__bf16*)carve((size_t)NNODES * 128 * 2);  // bf16 activations (ping)
  __bf16* hbB  = (__bf16*)carve((size_t)NNODES * 128 * 2);  // bf16 activations (pong)
  float*  t    = (float*)carve((size_t)NNODES * 128 * 4);   // GEMM output
  float*  agg  = (float*)carve((size_t)NNODES * 128 * 4);   // aggregation
  (void)ws_size; (void)n_in; (void)in_sizes; (void)out_size;

  const int B = 256;
  const int gemm_blocks = cdiv((long long)NNODES, 128);       // 8 M-tiles / block
  const int scat_blocks = cdiv((long long)NEDGES * 32, B);    // 1 wave / edge

  // ---- gcn_norm ----
  k_deg_init <<<cdiv(NNODES, B), B, 0, stream>>>(deg, NNODES);
  k_deg_accum<<<cdiv(NEDGES, B), B, 0, stream>>>(dst, ew, deg, NEDGES);
  k_dinv     <<<cdiv(NNODES, B), B, 0, stream>>>(deg, dinv, NNODES);
  k_norm     <<<cdiv(NEDGES, B), B, 0, stream>>>(src, dst, ew, dinv, norm, NEDGES);

  // ---- lin_in: h0 = relu(x @ W(128x64) + b), fused epilogue ----
  k_f32_to_bf16<<<cdiv((long long)NNODES * 128, B), B, 0, stream>>>(x, hbA, NNODES * 128);
  k_gemm_bf16<128, 64, 2><<<gemm_blocks, B, 0, stream>>>(
      hbA, lin_in_w, nullptr, hbB, nullptr, lin_in_b, nullptr, NNODES);

  // ---- conv1: 64 -> 128 ----
  k_gemm_bf16<64, 128, 1><<<gemm_blocks, B, 0, stream>>>(
      hbB, conv1_w, t, nullptr, dinv, conv1_b, agg, NNODES);
  k_scatter<128><<<scat_blocks, B, 0, stream>>>(t, src, dst, norm, agg, NEDGES);
  k_bn_relu_bf16<128><<<cdiv((long long)NNODES * 128, B), B, 0, stream>>>(
      agg, bn1_g, bn1_b, bn1_m, bn1_v, hbA, NNODES);

  // ---- conv2: 128 -> 128 ----
  k_gemm_bf16<128, 128, 1><<<gemm_blocks, B, 0, stream>>>(
      hbA, conv2_w, t, nullptr, dinv, conv2_b, agg, NNODES);
  k_scatter<128><<<scat_blocks, B, 0, stream>>>(t, src, dst, norm, agg, NEDGES);
  k_bn_relu_bf16<128><<<cdiv((long long)NNODES * 128, B), B, 0, stream>>>(
      agg, bn2_g, bn2_b, bn2_m, bn2_v, hbB, NNODES);

  // ---- conv3: 128 -> 64 ----
  k_gemm_bf16<128, 64, 1><<<gemm_blocks, B, 0, stream>>>(
      hbB, conv3_w, t, nullptr, dinv, conv3_b, agg, NNODES);
  k_scatter<64><<<scat_blocks, B, 0, stream>>>(t, src, dst, norm, agg, NEDGES);
  k_bn_relu_bf16<64><<<cdiv((long long)NNODES * 64, B), B, 0, stream>>>(
      agg, bn3_g, bn3_b, bn3_m, bn3_v, hbA, NNODES);

  // ---- lin_out: 64 -> 16, then log_softmax ----
  k_gemm_bf16<64, 16, 0><<<gemm_blocks, B, 0, stream>>>(
      hbA, lout_w, t, nullptr, nullptr, nullptr, nullptr, NNODES);
  k_logsoftmax16<<<cdiv(NNODES, B), B, 0, stream>>>(t, lout_b, out, NNODES);
}
